// Gemma4Attention_23210003268331
// MI455X (gfx1250) — matrix operand order repaired and checked by reading, not compiled, not run on hardware
//
#include <hip/hip_runtime.h>

// ---------------------------------------------------------------------------
// Gemma-4 attention block for MI455X (gfx1250, wave32, WMMA).
//   [0] f32->bf16 conversions; weights transposed to [n][k] (tiled via LDS)
//   [1] qkv = hidden @ w_qkv   (bf16 WMMA, f32 accum, dbl-buffered LDS,
//                               async global->LDS staging when available)
//   [2] RMSNorm + RoPE -> bf16 q/k/vT in WMMA-native layouts
//   [3] causal flash attention (bf16 WMMA, online softmax, bf16 out)
//   [4] out = attn @ w_o       (same GEMM)
// ---------------------------------------------------------------------------

static constexpr int Bc   = 2;
static constexpr int Sc   = 2048;
static constexpr int HIDc = 2048;
static constexpr int Hc   = 8;
static constexpr int HKVc = 4;
static constexpr int Dc   = 256;
static constexpr int Tc   = Bc * Sc;                 // 4096 tokens
static constexpr int QKVN = Hc * Dc + 2 * HKVc * Dc; // 4096 (q | k | v)

typedef __attribute__((ext_vector_type(16))) __bf16 v16bf;
typedef __attribute__((ext_vector_type(8)))  float  v8f;
typedef unsigned short     u16;
typedef unsigned long long u64;
typedef int v4i __attribute__((vector_size(16)));    // builtin's b128 type

// Async global->LDS copies (ASYNCcnt path). Guarded so both toolchains and
// the host pass still compile; fallback is a synchronous b128 copy.
#if defined(__HIP_DEVICE_COMPILE__) &&                                   \
    __has_builtin(__builtin_amdgcn_global_load_async_to_lds_b128) &&     \
    __has_builtin(__builtin_amdgcn_s_wait_asynccnt)
#define ASYNC_LDS 1
#else
#define ASYNC_LDS 0
#endif
#define AS1 __attribute__((address_space(1)))
#define AS3 __attribute__((address_space(3)))

union FragAB {                 // one WMMA A/B operand: 16 bf16 = 8 VGPRs
  v16bf v;
  u16   us[16];
  uint4 q[2];
};

__device__ inline u16 f2bf(float f) {              // f32 -> bf16 (RNE)
  unsigned int u = __float_as_uint(f);
  return (u16)((u + 0x7FFFu + ((u >> 16) & 1u)) >> 16);
}
__device__ inline u64 pack4bf(const float* x) {
  return (u64)f2bf(x[0]) | ((u64)f2bf(x[1]) << 16) |
         ((u64)f2bf(x[2]) << 32) | ((u64)f2bf(x[3]) << 48);
}
__device__ inline v8f wmma_bf16(const FragAB& a, const FragAB& b, v8f c) {
  return __builtin_amdgcn_wmma_f32_16x16x32_bf16(
      false, a.v, false, b.v, (short)0, c, false, false);
}
__device__ inline void copy_b128_to_lds(const u16* g, u16* l) {
#if ASYNC_LDS
  __builtin_amdgcn_global_load_async_to_lds_b128(
      (AS1 v4i*)g, (AS3 v4i*)l, 0, 0);
#else
  *(uint4*)l = *(const uint4*)g;
#endif
}

// ---------------------------------------------------------------------------
// [0a] streaming f32 -> bf16 (8 elements / thread), same layout
// ---------------------------------------------------------------------------
__global__ __launch_bounds__(256)
void f32_to_bf16(const float* __restrict__ in, u16* __restrict__ out, int n) {
  int i = (blockIdx.x * 256 + threadIdx.x) * 8;
  if (i >= n) return;
  float4 a = *(const float4*)(in + i);
  float4 b = *(const float4*)(in + i + 4);
  uint4 r;
  r.x = (unsigned)f2bf(a.x) | ((unsigned)f2bf(a.y) << 16);
  r.y = (unsigned)f2bf(a.z) | ((unsigned)f2bf(a.w) << 16);
  r.z = (unsigned)f2bf(b.x) | ((unsigned)f2bf(b.y) << 16);
  r.w = (unsigned)f2bf(b.z) | ((unsigned)f2bf(b.w) << 16);
  *(uint4*)(out + i) = r;
}

// ---------------------------------------------------------------------------
// [0b] f32 [K][N] -> bf16 transposed [N][K], 64x64 tiles through LDS so both
// the global read and the global write stay coalesced (one-time cost).
// ---------------------------------------------------------------------------
__global__ __launch_bounds__(256)
void f32_to_bf16_T(const float* __restrict__ in, u16* __restrict__ out,
                   int K, int N) {
  __shared__ __align__(16) u16 Ts[64 * 72];      // [n][k], padded stride
  const int kBase = blockIdx.y * 64;
  const int nBase = blockIdx.x * 64;
  const int tid = threadIdx.x;
#pragma unroll
  for (int q = 0; q < 4; ++q) {
    int idx = tid + q * 256;                     // [0,1024)
    int r = idx >> 4, c4 = (idx & 15) * 4;       // r = k row, c4 = n col
    float4 v = *(const float4*)(in + (size_t)(kBase + r) * N + nBase + c4);
    Ts[(c4 + 0) * 72 + r] = f2bf(v.x);
    Ts[(c4 + 1) * 72 + r] = f2bf(v.y);
    Ts[(c4 + 2) * 72 + r] = f2bf(v.z);
    Ts[(c4 + 3) * 72 + r] = f2bf(v.w);
  }
  __syncthreads();
#pragma unroll
  for (int q = 0; q < 2; ++q) {
    int idx = tid + q * 256;                     // [0,512)
    int n = idx >> 3, kc = (idx & 7) * 8;
    *(uint4*)(out + (size_t)(nBase + n) * K + kBase + kc) =
        *(const uint4*)&Ts[n * 72 + kc];
  }
}

// ---------------------------------------------------------------------------
// [1]/[4]  C[M,N](f32) = A[M,K](bf16) @ BT[N,K](bf16, pre-transposed).
// Block tile 128x128, 8 waves as 2x4; wave tile 64x32 = 4x2 WMMA tiles.
// Double-buffered LDS; both stages are contiguous b128 copies (async if
// the toolchain exposes GLOBAL_LOAD_ASYNC_TO_LDS).
// ---------------------------------------------------------------------------
#define BM 128
#define BN 128
#define BK 32

__global__ __launch_bounds__(256)
void gemm_bf16(const u16* __restrict__ A, const u16* __restrict__ BT,
               float* __restrict__ C, int M, int N, int K) {
  __shared__ __align__(16) u16 As[2][BM * BK];   // [m][k]
  __shared__ __align__(16) u16 Bs[2][BN * BK];   // [n][k]

  const int tid  = threadIdx.x;
  const int lane = tid & 31;
  const int wave = tid >> 5;
  const int wr = wave >> 2, wc = wave & 3;       // 2x4 wave grid
  const int rowBase = blockIdx.y * BM;
  const int colBase = blockIdx.x * BN;
  const int half = lane >> 4, l15 = lane & 15;

  auto stage = [&](int p, int k0) {
#pragma unroll
    for (int q = 0; q < 2; ++q) {                // A: 512 uint4, 2/thread
      int idx = tid + q * 256;
      int m = idx >> 2, kq = (idx & 3) * 8;
      copy_b128_to_lds(A + (size_t)(rowBase + m) * K + k0 + kq,
                       &As[p][m * BK + kq]);
    }
#pragma unroll
    for (int q = 0; q < 2; ++q) {                // B: 512 uint4, 2/thread
      int idx = tid + q * 256;
      int n = idx >> 2, kq = (idx & 3) * 8;
      copy_b128_to_lds(BT + (size_t)(colBase + n) * K + k0 + kq,
                       &Bs[p][n * BK + kq]);
    }
  };

  v8f acc[4][2];
  const v8f vzero = {0.f, 0.f, 0.f, 0.f, 0.f, 0.f, 0.f, 0.f};
#pragma unroll
  for (int i = 0; i < 4; ++i)
#pragma unroll
    for (int j = 0; j < 2; ++j) acc[i][j] = vzero;

  stage(0, 0);
#if ASYNC_LDS
  __builtin_amdgcn_s_wait_asynccnt(0);
#endif
  __syncthreads();
  int p = 0;
  for (int k0 = 0; k0 < K; k0 += BK) {
    if (k0 + BK < K) stage(p ^ 1, k0 + BK);      // overlap with compute
    if (k0 + 2 * BK < K)                          // L2 prefetch, tile after next
      __builtin_prefetch(A + (size_t)(rowBase + (tid >> 1)) * K + k0 + 2 * BK, 0, 1);

    FragAB af[4];
    const int kbA = half * 16;
#pragma unroll
    for (int i = 0; i < 4; ++i) {
      int rl = wr * 64 + i * 16 + l15;
      af[i].q[0] = *(const uint4*)&As[p][rl * BK + kbA];
      af[i].q[1] = *(const uint4*)&As[p][rl * BK + kbA + 8];
    }
    FragAB bf[2];
    const int kbB = half * 16;
#pragma unroll
    for (int j = 0; j < 2; ++j) {
      int cl = wc * 32 + j * 16 + l15;
      bf[j].q[0] = *(const uint4*)&Bs[p][cl * BK + kbB];
      bf[j].q[1] = *(const uint4*)&Bs[p][cl * BK + kbB + 8];
    }
#pragma unroll
    for (int i = 0; i < 4; ++i)
#pragma unroll
      for (int j = 0; j < 2; ++j)
        acc[i][j] = wmma_bf16(af[i], bf[j], acc[i][j]);
#if ASYNC_LDS
    __builtin_amdgcn_s_wait_asynccnt(0);         // our async writes landed
#endif
    __syncthreads();                             // everyone's landed
    p ^= 1;
  }

#pragma unroll
  for (int i = 0; i < 4; ++i)
#pragma unroll
    for (int j = 0; j < 2; ++j)
#pragma unroll
      for (int r = 0; r < 8; ++r) {
        int row = rowBase + wr * 64 + i * 16 + r + 8 * half;
        int col = colBase + wc * 32 + j * 16 + l15;
        C[(size_t)row * N + col] = acc[i][j][r];
      }
}

// ---------------------------------------------------------------------------
// [2] RMSNorm (+weight q/k) + RoPE (q/k), f32 qkv -> bf16 q/k/vT buffers.
// One wave per (token, slot); slot 0-7 = q heads, 8-11 = k, 12-15 = v.
// Lane owns d = lane*4+{0..3} and d+128 -> RoPE pair is lane-local.
// Layouts: qb[t][h*256+d], kb[(b*HKV+kvh)*S + s][d], vT[(b*HKV+kvh)*256+d][s].
// ---------------------------------------------------------------------------
__global__ __launch_bounds__(256)
void rmsnorm_rope(const float* __restrict__ qkvf, const int* __restrict__ positions,
                  const float* __restrict__ qw, const float* __restrict__ kw,
                  u16* __restrict__ qb, u16* __restrict__ kbuf, u16* __restrict__ vT) {
  const int gw   = blockIdx.x * 8 + (threadIdx.x >> 5);
  const int lane = threadIdx.x & 31;
  const int t = gw >> 4, slot = gw & 15;
  const int b = t >> 11, s = t & (Sc - 1);

  const float* src;
  const float* w = nullptr;
  bool doRope = true;
  const size_t row = (size_t)t * QKVN;
  if (slot < 8)       { src = qkvf + row + slot * Dc;                      w = qw; }
  else if (slot < 12) { src = qkvf + row + Hc * Dc + (slot - 8) * Dc;      w = kw; }
  else                { src = qkvf + row + Hc * Dc + HKVc * Dc + (slot - 12) * Dc; doRope = false; }

  const int d0 = lane * 4;
  float4 a = *(const float4*)(src + d0);
  float4 bv = *(const float4*)(src + d0 + 128);
  float x1[4] = {a.x, a.y, a.z, a.w};
  float x2[4] = {bv.x, bv.y, bv.z, bv.w};

  float ss = 0.f;
#pragma unroll
  for (int j = 0; j < 4; ++j) ss += x1[j] * x1[j] + x2[j] * x2[j];
#pragma unroll
  for (int m = 16; m >= 1; m >>= 1) ss += __shfl_xor(ss, m, 32);
  const float inv = rsqrtf(ss * (1.0f / Dc) + 1e-6f);

#pragma unroll
  for (int j = 0; j < 4; ++j) {
    float w1 = w ? (1.0f + w[d0 + j])       : 1.0f;
    float w2 = w ? (1.0f + w[d0 + 128 + j]) : 1.0f;
    x1[j] *= inv * w1;
    x2[j] *= inv * w2;
  }
  if (doRope) {
    const float p = (float)positions[t];
#pragma unroll
    for (int j = 0; j < 4; ++j) {
      float fr = __powf(10000.0f, -(float)(d0 + j) * (1.0f / 128.0f));
      float sn, cs;
      __sincosf(p * fr, &sn, &cs);
      float o1 = x1[j] * cs - x2[j] * sn;
      float o2 = x2[j] * cs + x1[j] * sn;
      x1[j] = o1; x2[j] = o2;
    }
  }

  if (slot < 8) {                      // q: [t][h*256+d]
    u16* dst = qb + (size_t)t * (Hc * Dc) + slot * Dc;
    *(u64*)(dst + d0)       = pack4bf(x1);
    *(u64*)(dst + d0 + 128) = pack4bf(x2);
  } else if (slot < 12) {              // k: [(b*HKV+kvh)*S + s][d]
    u16* dst = kbuf + ((size_t)(b * HKVc + (slot - 8)) * Sc + s) * Dc;
    *(u64*)(dst + d0)       = pack4bf(x1);
    *(u64*)(dst + d0 + 128) = pack4bf(x2);
  } else {                             // v transposed: [(b*HKV+kvh)*256 + d][s]
    u16* dst = vT + (size_t)(b * HKVc + (slot - 12)) * Dc * Sc + s;
#pragma unroll
    for (int j = 0; j < 4; ++j) {
      dst[(size_t)(d0 + j) * Sc]       = f2bf(x1[j]);
      dst[(size_t)(d0 + 128 + j) * Sc] = f2bf(x2[j]);
    }
  }
}

// ---------------------------------------------------------------------------
// [3] Causal flash attention, all-bf16 operands. One wave per 16-row Q tile;
// 32-key chunks; online softmax; bf16 output. No block barrier (causal trip
// counts differ per wave).
// ---------------------------------------------------------------------------
__global__ __launch_bounds__(256)
void flash_attn(const u16* __restrict__ qb, const u16* __restrict__ kbuf,
                const u16* __restrict__ vT, u16* __restrict__ attnb) {
  __shared__ __align__(16) u16 Ps[8][16 * 32];   // per-wave P transpose

  const int lane = threadIdx.x & 31;
  const int wave = threadIdx.x >> 5;
  const int qblocks = Sc / 128;                  // 16
  const int bh   = blockIdx.x / qblocks;
  const int qblk = blockIdx.x % qblocks;
  const int b = bh >> 3, h = bh & 7;
  const int kvh = h >> 1;                        // H/HKV = 2
  const int qtile = qblk * 8 + wave;
  const int half = lane >> 4, l15 = lane & 15;

  const u16* kBase = kbuf + (size_t)(b * HKVc + kvh) * Sc * Dc;
  const u16* vBase = vT   + (size_t)(b * HKVc + kvh) * Dc * Sc;

  // ---- Q fragments (A-layout): two b128 loads per fragment
  FragAB qf[8];
  {
    const int tq = b * Sc + qtile * 16 + l15;
    const u16* qp = qb + (size_t)tq * (Hc * Dc) + h * Dc + half * 16;
#pragma unroll
    for (int db = 0; db < 8; ++db) {
      qf[db].q[0] = *(const uint4*)(qp + db * 32);
      qf[db].q[1] = *(const uint4*)(qp + db * 32 + 8);
    }
  }

  const v8f vzero = {0.f, 0.f, 0.f, 0.f, 0.f, 0.f, 0.f, 0.f};
  v8f o[16];
#pragma unroll
  for (int t = 0; t < 16; ++t) o[t] = vzero;
  float m_[8], l_[8];
#pragma unroll
  for (int r = 0; r < 8; ++r) { m_[r] = -3.0e38f; l_[r] = 0.f; }

  const int qrow0 = qtile * 16 + 8 * half;
  const int nChunks = (qtile * 16 + 47) >> 5;    // ceil((qmax+1)/32)

  for (int ch = 0; ch < nChunks; ++ch) {
    const int kb32 = ch * 32;
    float s0[8], s1[8];

    // ---- scores: two 16x16 tiles, 8 WMMAs each over D=256
#pragma unroll
    for (int sc = 0; sc < 2; ++sc) {
      v8f acc = vzero;
      const int key = kb32 + sc * 16 + l15;      // this lane's key column
      const u16* kp = kBase + (size_t)key * Dc + half * 16;
#pragma unroll
      for (int db = 0; db < 8; ++db) {
        FragAB kf;
        kf.q[0] = *(const uint4*)(kp + db * 32);
        kf.q[1] = *(const uint4*)(kp + db * 32 + 8);
        acc = wmma_bf16(qf[db], kf, acc);
      }
      float* sp = sc ? s1 : s0;
#pragma unroll
      for (int r = 0; r < 8; ++r)
        sp[r] = (key > qrow0 + r) ? -3.0e38f : acc[r];   // causal mask
    }

    // ---- online softmax: row stats across 16-lane halves (masks 8..1)
    float pscale[8];
#pragma unroll
    for (int r = 0; r < 8; ++r) {
      float mx = fmaxf(s0[r], s1[r]);
#pragma unroll
      for (int msk = 8; msk >= 1; msk >>= 1) mx = fmaxf(mx, __shfl_xor(mx, msk, 32));
      float mn = fmaxf(m_[r], mx);
      pscale[r] = __expf(m_[r] - mn);
      m_[r] = mn;
      s0[r] = __expf(s0[r] - mn);
      s1[r] = __expf(s1[r] - mn);
      float rs = s0[r] + s1[r];
#pragma unroll
      for (int msk = 8; msk >= 1; msk >>= 1) rs += __shfl_xor(rs, msk, 32);
      l_[r] = l_[r] * pscale[r] + rs;
    }
#pragma unroll
    for (int t = 0; t < 16; ++t)
#pragma unroll
      for (int r = 0; r < 8; ++r) o[t][r] *= pscale[r];

    // ---- transpose P (C-layout -> A-layout) through per-wave LDS
#pragma unroll
    for (int r = 0; r < 8; ++r) {
      int rowl = r + 8 * half;
      Ps[wave][rowl * 32 + l15]      = f2bf(s0[r]);
      Ps[wave][rowl * 32 + 16 + l15] = f2bf(s1[r]);
    }
    __builtin_amdgcn_wave_barrier();
    FragAB pa;
    pa.q[0] = *(const uint4*)&Ps[wave][l15 * 32 + half * 16];
    pa.q[1] = *(const uint4*)&Ps[wave][l15 * 32 + half * 16 + 8];

    // ---- O += P @ V : V fragment = two contiguous b128 loads from vT
#pragma unroll
    for (int dt = 0; dt < 16; ++dt) {
      FragAB vf;
      const u16* vp = vBase + (size_t)(dt * 16 + l15) * Sc + kb32 + half * 16;
      vf.q[0] = *(const uint4*)(vp);
      vf.q[1] = *(const uint4*)(vp + 8);
      o[dt] = wmma_bf16(pa, vf, o[dt]);
    }
  }

  // ---- normalize, convert, store bf16
  float invl[8];
#pragma unroll
  for (int r = 0; r < 8; ++r) invl[r] = 1.0f / l_[r];
#pragma unroll
  for (int dt = 0; dt < 16; ++dt)
#pragma unroll
    for (int r = 0; r < 8; ++r) {
      int trow = b * Sc + qtile * 16 + r + 8 * half;
      attnb[(size_t)trow * (Hc * Dc) + h * Dc + dt * 16 + l15] =
          f2bf(o[dt][r] * invl[r]);
    }
}

// ---------------------------------------------------------------------------
extern "C" void kernel_launch(void* const* d_in, const int* in_sizes, int n_in,
                              void* d_out, int out_size, void* d_ws, size_t ws_size,
                              hipStream_t stream) {
  const float* hidden    = (const float*)d_in[0];   // [T, HID]
  const int*   positions = (const int*)  d_in[1];   // [T]
  const float* w_qkv     = (const float*)d_in[2];   // [HID, 4096]
  const float* q_norm_w  = (const float*)d_in[3];   // [256]
  const float* k_norm_w  = (const float*)d_in[4];   // [256]
  const float* w_o       = (const float*)d_in[5];   // [2048, 2048]

  // workspace layout (kernels serialize on stream; regions alias over time):
  //   [  0, 64) MB : qkv f32 (GEMM1 out, rmsnorm in)
  //   [ 64, 80) MB : hid_bf   -> reused as qb    after GEMM1
  //   [ 80, 96) MB : wqkvT_bf -> reused as kb|vT after GEMM1
  //   [ 96,112) MB : attn bf16
  //   [112,120) MB : w_o^T bf16
  char* ws = (char*)d_ws;
  float* qkvf  = (float*)ws;
  u16* hid_bf  = (u16*)(ws + (size_t)64  * (1 << 20));
  u16* wqkvT   = (u16*)(ws + (size_t)80  * (1 << 20));
  u16* qb      = (u16*)(ws + (size_t)64  * (1 << 20));
  u16* kb      = (u16*)(ws + (size_t)80  * (1 << 20));
  u16* vT      = (u16*)(ws + (size_t)88  * (1 << 20));
  u16* attnb   = (u16*)(ws + (size_t)96  * (1 << 20));
  u16* woT     = (u16*)(ws + (size_t)112 * (1 << 20));

  // [0] operand conversions (weights transposed to [n][k])
  f32_to_bf16<<<Tc * HIDc / 2048, 256, 0, stream>>>(hidden, hid_bf, Tc * HIDc);
  f32_to_bf16_T<<<dim3(QKVN / 64, HIDc / 64), 256, 0, stream>>>(w_qkv, wqkvT, HIDc, QKVN);
  f32_to_bf16_T<<<dim3(HIDc / 64, HIDc / 64), 256, 0, stream>>>(w_o, woT, HIDc, HIDc);
  // [1] qkv = hidden @ w_qkv
  gemm_bf16<<<dim3(QKVN / BN, Tc / BM), 256, 0, stream>>>(
      hid_bf, wqkvT, qkvf, Tc, QKVN, HIDc);
  // [2] RMSNorm + RoPE -> bf16 q / k / vT
  rmsnorm_rope<<<Tc * 16 / 8, 256, 0, stream>>>(
      qkvf, positions, q_norm_w, k_norm_w, qb, kb, vT);
  // [3] causal flash attention -> attn bf16
  flash_attn<<<Bc * Hc * (Sc / 128), 256, 0, stream>>>(qb, kb, vT, attnb);
  // [4] out = attn @ w_o
  gemm_bf16<<<dim3(HIDc / BN, Tc / BM), 256, 0, stream>>>(
      attnb, woT, (float*)d_out, Tc, HIDc, Hc * Dc);
}
